// Block_67302137528539
// MI455X (gfx1250) — matrix operand order repaired and checked by reading, not compiled, not run on hardware
//
#include <hip/hip_runtime.h>
#include <hip/hip_bf16.h>
#include <cstdint>

#define DI __device__ __forceinline__

typedef unsigned short u16;
typedef __attribute__((ext_vector_type(16))) __bf16 bf16x16;
typedef __attribute__((ext_vector_type(8)))  float  f32x8;
typedef int v4i __attribute__((vector_size(16)));

union BFrag { bf16x16 v; uint4 q[2]; u16 s[16]; };

static constexpr int B_ = 8, C_ = 512, W_ = 64, H_ = 32, HID_ = 256, G_ = 1280;
static constexpr int M_   = W_ * H_ * B_;        // 16384 rows of the big GEMM
static constexpr int PW   = W_ + 1, PH = H_ + 1; // zero-padded halo grid 65 x 33
static constexpr int CELL = B_ * HID_;           // 2048 elems per padded (w,h) slot
static constexpr int NOUT = B_ * HID_ * W_ * H_; // 4194304 per output tensor
static constexpr int NDIAG = W_ + H_ - 1;        // 95 wavefront steps
static constexpr unsigned NBLK_REC = 32;         // persistent recurrence blocks

// ---- gfx1250 async-to-LDS path (guarded: falls back to sync copies if absent)
#if defined(__has_builtin)
#if __has_builtin(__builtin_amdgcn_global_load_async_to_lds_b128) && \
    __has_builtin(__builtin_amdgcn_s_wait_asynccnt)
#define USE_ASYNC_LDS 1
#endif
#endif
#ifndef USE_ASYNC_LDS
#define USE_ASYNC_LDS 0
#endif

#if USE_ASYNC_LDS
typedef __attribute__((address_space(1))) v4i gv4i;   // global b128 payload
typedef __attribute__((address_space(3))) v4i lv4i;   // LDS b128 payload
DI void async_cp16(const u16* g, u16* l) {   // 16B per lane, ASYNCcnt-tracked
  __builtin_amdgcn_global_load_async_to_lds_b128((gv4i*)g, (lv4i*)l, 0, 0);
}
#define ASYNC_WAIT() __builtin_amdgcn_s_wait_asynccnt(0)
#else
DI void async_cp16(const u16* g, u16* l) { *(uint4*)l = *(const uint4*)g; }
#define ASYNC_WAIT() ((void)0)
#endif

DI u16 f2bf(float f) {                          // round-to-nearest-even fp32 -> bf16
  unsigned u = __float_as_uint(f);
  unsigned r = u + 0x7FFFu + ((u >> 16) & 1u);
  return (u16)(r >> 16);
}
DI float sigm(float x) { return 1.0f / (1.0f + __expf(-x)); }

// device-wide barrier for the persistent recurrence kernel (32 resident blocks)
DI void grid_sync(unsigned* bar) {
  __syncthreads();
  if (threadIdx.x == 0) {
    unsigned* cnt = bar;
    unsigned* gen = bar + 1;
    __threadfence();                            // release h/c writes of this block
    unsigned g = __atomic_load_n(gen, __ATOMIC_ACQUIRE);
    if (atomicAdd(cnt, 1u) == NBLK_REC - 1) {
      __atomic_store_n(cnt, 0u, __ATOMIC_RELAXED);
      __atomic_fetch_add(gen, 1u, __ATOMIC_RELEASE);
    } else {
      while (__atomic_load_n(gen, __ATOMIC_ACQUIRE) == g)
        __builtin_amdgcn_s_sleep(2);
    }
    __threadfence();                            // acquire other blocks' writes
  }
  __syncthreads();
}

// ---------------------------------------------------------------- utilities
__global__ void k_zero(uint32_t* p, int n) {
  int i = blockIdx.x * 256 + threadIdx.x;
  if (i < n) p[i] = 0u;
}

// x (B,C,W,H) f32  ->  xT[(w,h,b), c] bf16   (row-major, K-contiguous for WMMA A)
__global__ void k_xT(const float* __restrict__ x, u16* __restrict__ xT) {
  int idx = blockIdx.x * 256 + threadIdx.x;     // over M_*C_
  int c = idx & (C_ - 1);
  int whb = idx >> 9;
  int w = whb >> 8, h = (whb >> 3) & 31, b = whb & 7;
  float v = x[(((size_t)b * C_ + c) * W_ + w) * H_ + h];
  xT[idx] = f2bf(v);
}

// Wx (C,G) -> wxT[n][k] bf16: column-major so B-tiles stage as pure b128 copies
__global__ void k_wxT(const float* __restrict__ wx, u16* __restrict__ wxT) {
  int i = blockIdx.x * 256 + threadIdx.x;       // over G_*C_
  int n = i >> 9, k = i & 511;
  wxT[i] = f2bf(wx[(size_t)k * G_ + n]);
}

// whT[n][k] bf16, k in [0,512): k<256 -> Wh1[k][n], else Wh2[k-256][n]
// (K-concatenated so  h_l@Wh1 + h_u@Wh2  ==  [h_l ; h_u] @ whT^T  in ONE GEMM)
__global__ void k_whT(const float* __restrict__ wh1, const float* __restrict__ wh2,
                      u16* __restrict__ whT) {
  int i = blockIdx.x * 256 + threadIdx.x;       // over G_*512
  int n = i >> 9, k = i & 511;
  float v = (k < HID_) ? wh1[(size_t)k * G_ + n] : wh2[(size_t)(k - HID_) * G_ + n];
  whT[i] = f2bf(v);
}

// ---------------------------------------------- Phase 1: xp = xT @ Wx + bias
// M=16384, N=1280, K=512; block tile 128x64, 8 waves, each wave 16x64 (4 WMMA
// tiles). Double-buffered LDS; tiles stream in via GLOBAL_LOAD_ASYNC_TO_LDS_B128
// while the current buffer feeds v_wmma.
__global__ __launch_bounds__(256) void k_gemm1(const u16* __restrict__ xT,
                                               const u16* __restrict__ wxT,
                                               const float* __restrict__ bias,
                                               float* __restrict__ xp) {
  __shared__ u16 As[2][128 * 32];  // A tile row-major [m][k]
  __shared__ u16 Bs[2][64 * 32];   // B tile [n][k] (wxT already transposed)
  const int tid = threadIdx.x, lane = tid & 31, wv = tid >> 5;
  const int m0 = blockIdx.x * 128, n0 = blockIdx.y * 64;

  f32x8 acc[4];
#pragma unroll
  for (int t = 0; t < 4; ++t)
#pragma unroll
    for (int j = 0; j < 8; ++j) acc[t][j] = 0.0f;

  // staging map: A = 128x32 u16 (2 threads/row, 16 u16 each = 2 x b128)
  //              B =  64x32 u16 (4 threads/row,  8 u16 each = 1 x b128)
  const int ar = tid >> 1, ak = (tid & 1) * 16;
  const int bn = tid >> 2, bkof = (tid & 3) * 8;
  const u16* gA = xT + (size_t)(m0 + ar) * 512 + ak;
  const u16* gB = wxT + (size_t)(n0 + bn) * 512 + bkof;
  u16* lA0 = &As[0][ar * 32 + ak];
  u16* lB0 = &Bs[0][bn * 32 + bkof];
  const int lstride = 128 * 32;    // elems between the two A buffers
  const int lstrideB = 64 * 32;

  auto stage = [&](int k0, int buf) {
    async_cp16(gA + k0,     lA0 + buf * lstride);
    async_cp16(gA + k0 + 8, lA0 + buf * lstride + 8);
    async_cp16(gB + k0,     lB0 + buf * lstrideB);
  };

  const int frow = wv * 16 + (lane & 15);
  const int khalf = lane >> 4;

  stage(0, 0);
  ASYNC_WAIT();
  __syncthreads();

  int buf = 0;
  for (int ks = 0; ks < 16; ++ks) {
    if (ks < 15) stage((ks + 1) * 32, buf ^ 1);   // prefetch next tile into other buffer

    // A frag: lane<16 holds K {0-7,16-23}, lane>=16 K {8-15,24-31} (ISA layout)
    BFrag a;
    a.q[0] = *(const uint4*)(&As[buf][frow * 32 + khalf * 8]);
    a.q[1] = *(const uint4*)(&As[buf][frow * 32 + 16 + khalf * 8]);
#pragma unroll
    for (int t = 0; t < 4; ++t) {
      BFrag bf;
      const u16* bp = &Bs[buf][(t * 16 + (lane & 15)) * 32 + khalf * 8];
      bf.q[0] = *(const uint4*)(bp);
      bf.q[1] = *(const uint4*)(bp + 16);
      acc[t] = __builtin_amdgcn_wmma_f32_16x16x32_bf16(false, a.v, false, bf.v,
                                                       (short)0, acc[t], false, false);
    }
    ASYNC_WAIT();        // my async writes into buf^1 landed
    __syncthreads();     // everyone done reading buf / staging buf^1
    buf ^= 1;
  }

  const int mhi = (lane < 16) ? 0 : 8;  // C layout: VGPR i -> row i / i+8, lane -> col
#pragma unroll
  for (int t = 0; t < 4; ++t) {
    int col = n0 + t * 16 + (lane & 15);
    float bb = bias[col];
#pragma unroll
    for (int i = 0; i < 8; ++i) {
      int row = m0 + wv * 16 + mhi + i;
      xp[(size_t)row * G_ + col] = acc[t][i] + bb;
    }
  }
}

// ----------------------- Phase 2: persistent wavefront 2D-LSTM recurrence
// Fixed grid of 32 blocks = 16 cell-pairs x 2 hid-halves (max anti-diagonal
// parallelism). Loops over all 95 diagonals with a device-wide barrier —
// removes 95 kernel-launch overheads. Per pair: 2 cells packed into the 16
// WMMA M-rows; each wave owns hid range [yh*128 + wv*16, +16) and computes the
// 5 gate tiles itself, so the nonlinear update stays in accumulator registers.
__global__ __launch_bounds__(256) void k_rec(
    const float* __restrict__ xp, const u16* __restrict__ whT,
    u16* __restrict__ hpad, float* __restrict__ cpad, float* __restrict__ state,
    unsigned* bar) {
  const int tid = threadIdx.x, lane = tid & 31, wv = tid >> 5;
  const int pair = blockIdx.x & 15;
  const int yh = blockIdx.x >> 4;

  const int lane15 = lane & 15;
  const int khalf = lane >> 4;
  const int cellA = lane15 >> 3;   // which cell this A-row belongs to
  const int ba = lane15 & 7;       // batch index within the cell
  const int hidb = yh * 128 + wv * 16;
  const int hid = hidb + lane15;
  const int mhi = (lane < 16) ? 0 : 8;

  for (int d = 0; d < NDIAG; ++d) {
    const int hmin = max(0, d - (W_ - 1));
    const int hmax = min(H_ - 1, d);
    const int cnt = hmax - hmin + 1;
    const int j0 = pair * 2;
    if (j0 < cnt) {                               // block-uniform: EXEC stays full
      const int j1 = min(j0 + 1, cnt - 1);        // clamp -> safe math, guarded store
      const bool valid1 = (j0 + 1) < cnt;
      const int h0 = hmin + j0, w0 = d - h0;
      const int h1 = hmin + j1, w1 = d - h1;

      // padded halo: cell (w,h) lives at (w+1,h+1); borders are zero
      const u16* hL0 = hpad + ((size_t)(w0 + 1) * PH + h0) * CELL;   // (w0,h0-1)
      const u16* hU0 = hpad + ((size_t)w0 * PH + (h0 + 1)) * CELL;   // (w0-1,h0)
      const u16* hL1 = hpad + ((size_t)(w1 + 1) * PH + h1) * CELL;
      const u16* hU1 = hpad + ((size_t)w1 * PH + (h1 + 1)) * CELL;
      const u16* hl = cellA ? hL1 : hL0;
      const u16* hu = cellA ? hU1 : hU0;
      const int r0 = (w0 * H_ + h0) * B_;
      const int r1 = (w1 * H_ + h1) * B_;

      // C initialized from precomputed xp (gate bias already folded in)
      f32x8 acc[5];
#pragma unroll
      for (int g = 0; g < 5; ++g) {
        int col = g * HID_ + hid;
#pragma unroll
        for (int i = 0; i < 8; ++i) {
          int m = mhi + i;
          int xr = ((m < 8) ? r0 : r1) + (m & 7);
          acc[g][i] = xp[(size_t)xr * G_ + col];
        }
      }

      // K = 512 = [h_left(256) ; h_up(256)] : 16 WMMA K-steps x 5 gate tiles
      for (int ks = 0; ks < 16; ++ks) {
        const int kb = ks * 32;
        BFrag a;
        {
          int kk = kb + khalf * 8;
          const u16* p = (kk < HID_) ? (hl + ba * HID_ + kk)
                                     : (hu + ba * HID_ + (kk - HID_));
          a.q[0] = *(const uint4*)p;
        }
        {
          int kk = kb + 16 + khalf * 8;
          const u16* p = (kk < HID_) ? (hl + ba * HID_ + kk)
                                     : (hu + ba * HID_ + (kk - HID_));
          a.q[1] = *(const uint4*)p;
        }
#pragma unroll
        for (int g = 0; g < 5; ++g) {
          int col = g * HID_ + hid;
          const u16* bp = whT + (size_t)col * 512 + kb + khalf * 8;
          BFrag bf;
          bf.q[0] = *(const uint4*)bp;
          bf.q[1] = *(const uint4*)(bp + 16);
          acc[g] = __builtin_amdgcn_wmma_f32_16x16x32_bf16(false, a.v, false, bf.v,
                                                           (short)0, acc[g], false, false);
        }
      }

      // LSTM cell update entirely in-register: all 5 gates of (m,hid) in this lane
#pragma unroll
      for (int i = 0; i < 8; ++i) {
        int m = mhi + i;
        int cell = m >> 3;
        if (cell && !valid1) continue;
        int b = m & 7;
        int w = cell ? w1 : w0;
        int h = cell ? h1 : h0;
        float iv  = sigm(acc[0][i]);
        float f1v = sigm(acc[1][i]);
        float f2v = sigm(acc[2][i]);
        float ov  = sigm(acc[3][i]);
        float cd  = tanhf(acc[4][i]);
        size_t lo = (size_t)b * HID_ + hid;
        float cl = cpad[((size_t)(w + 1) * PH + h) * CELL + lo];
        float cu = cpad[((size_t)w * PH + (h + 1)) * CELL + lo];
        float c  = iv * cd + f1v * cl + f2v * cu;
        float hv = ov * tanhf(c);
        size_t pc = ((size_t)(w + 1) * PH + (h + 1)) * CELL + lo;
        cpad[pc] = c;
        hpad[pc] = f2bf(hv);                       // feeds next diagonal's WMMA A
        state[((size_t)(w * H_ + h) * B_ + b) * HID_ + hid] = hv;
      }
    }
    grid_sync(bar);                                // diagonal d visible everywhere
  }
}

// ------------------------------------------- Phase 3: BatchNorm stats + finalize
__global__ __launch_bounds__(256) void k_bnred(const float* __restrict__ state,
                                               float* __restrict__ bn) {
  int t = threadIdx.x;                       // channel
  int rb = blockIdx.x * 128;
  float s = 0.0f, s2 = 0.0f;
  for (int r = 0; r < 128; ++r) {
    float v = state[(size_t)(rb + r) * HID_ + t];
    s += v; s2 += v * v;
  }
  atomicAdd(&bn[t], s);
  atomicAdd(&bn[HID_ + t], s2);
}

__global__ __launch_bounds__(256) void k_final(const float* __restrict__ state,
                                               const float* __restrict__ cpad,
                                               const float* __restrict__ bn,
                                               const float* __restrict__ gamma,
                                               const float* __restrict__ beta,
                                               float* __restrict__ out) {
  int idx = blockIdx.x * 256 + threadIdx.x;  // (B,HID,W,H) flat
  int h = idx & 31;
  int w = (idx >> 5) & 63;
  int ch = (idx >> 11) & 255;
  int b = idx >> 19;
  float s = state[((size_t)(w * H_ + h) * B_ + b) * HID_ + ch];
  float c = cpad[((size_t)(w + 1) * PH + (h + 1)) * CELL + (size_t)b * HID_ + ch];
  float mean = bn[ch] * (1.0f / (float)M_);
  float var  = bn[HID_ + ch] * (1.0f / (float)M_) - mean * mean;
  float xn = (s - mean) * rsqrtf(var + 1e-5f);
  out[idx] = tanhf(xn * gamma[ch] + beta[ch]);
  out[NOUT + idx] = s;
  out[2 * NOUT + idx] = c;
}

// ------------------------------------------------------------------ launcher
extern "C" void kernel_launch(void* const* d_in, const int* in_sizes, int n_in,
                              void* d_out, int out_size, void* d_ws, size_t ws_size,
                              hipStream_t stream) {
  (void)in_sizes; (void)n_in; (void)out_size; (void)ws_size;
  const float* x     = (const float*)d_in[0];
  const float* Wx    = (const float*)d_in[1];
  const float* Wh1   = (const float*)d_in[2];
  const float* Wh2   = (const float*)d_in[3];
  const float* bias  = (const float*)d_in[4];
  const float* gamma = (const float*)d_in[5];
  const float* beta  = (const float*)d_in[6];
  float* out = (float*)d_out;

  char* ws = (char*)d_ws;
  size_t off = 0;
  auto alloc = [&](size_t bytes) -> char* {
    char* p = ws + off;
    off = (off + bytes + 255) & ~(size_t)255;
    return p;
  };
  u16*   xT    = (u16*)  alloc((size_t)M_ * C_ * 2);        // 16 MB
  u16*   wxT   = (u16*)  alloc((size_t)G_ * 512 * 2);       // 1.3 MB
  u16*   whT   = (u16*)  alloc((size_t)G_ * 512 * 2);       // 1.3 MB (L2-resident)
  float* xp    = (float*)alloc((size_t)M_ * G_ * 4);        // 84 MB
  u16*   hpad  = (u16*)  alloc((size_t)PW * PH * CELL * 2); // 8.8 MB
  float* cpad  = (float*)alloc((size_t)PW * PH * CELL * 4); // 17.6 MB
  float* state = (float*)alloc((size_t)M_ * HID_ * 4);      // 16.8 MB
  float* bn    = (float*)alloc((size_t)2 * HID_ * 4);
  unsigned* bar = (unsigned*)alloc(2 * sizeof(unsigned));

  // zero halo buffers + BN accumulators + barrier (every call: no re-poison)
  int nz1 = PW * PH * CELL / 2;  // hpad as u32 words
  k_zero<<<(nz1 + 255) / 256, 256, 0, stream>>>((uint32_t*)hpad, nz1);
  int nz2 = PW * PH * CELL;
  k_zero<<<(nz2 + 255) / 256, 256, 0, stream>>>((uint32_t*)cpad, nz2);
  k_zero<<<2, 256, 0, stream>>>((uint32_t*)bn, 2 * HID_);
  k_zero<<<1, 256, 0, stream>>>((uint32_t*)bar, 2);

  k_xT <<<(M_ * C_) / 256, 256, 0, stream>>>(x, xT);
  k_wxT<<<(G_ * 512) / 256, 256, 0, stream>>>(Wx, wxT);
  k_whT<<<(G_ * 512) / 256, 256, 0, stream>>>(Wh1, Wh2, whT);

  k_gemm1<<<dim3(M_ / 128, G_ / 64), 256, 0, stream>>>(xT, wxT, bias, xp);

  // single persistent kernel walks all 95 anti-diagonals with device-wide sync
  k_rec<<<NBLK_REC, 256, 0, stream>>>(xp, whT, hpad, cpad, state, bar);

  k_bnred<<<M_ / 128, 256, 0, stream>>>(state, bn);
  k_final<<<NOUT / 256, 256, 0, stream>>>(state, cpad, bn, gamma, beta, out);
}
